// GraphAttentionHead_10754598109593
// MI455X (gfx1250) — matrix operand order repaired and checked by reading, not compiled, not run on hardware
//
#include <hip/hip_runtime.h>

typedef __attribute__((ext_vector_type(16))) __bf16 v16bf;
typedef __attribute__((ext_vector_type(8)))  __bf16 v8bf;
typedef __attribute__((ext_vector_type(8)))  float  v8f;

#define ALPHA 0.2f
#define NROWS 16384        // B*N = 8*2048
#define NN    2048
#define FIN   512
#define FOUT  256

// ---------------------------------------------------------------------------
// Kernel 0a: zero Wh1/Wh2 accumulators (graph-safe, runs every launch)
// ---------------------------------------------------------------------------
__global__ __launch_bounds__(256)
void zero_init(float* __restrict__ p, int n)
{
    const int i = blockIdx.x * 1024 + threadIdx.x;
    #pragma unroll
    for (int t = 0; t < 4; ++t)
        if (i + t * 256 < n) p[i + t * 256] = 0.f;
}

// ---------------------------------------------------------------------------
// Kernel 0b: Wt[col][k] = bf16(W[k][col])  — transposed bf16 weights so the
// projection GEMM's B-fragments are contiguous 32-byte loads.
// ---------------------------------------------------------------------------
__global__ __launch_bounds__(256)
void prep_wt(const float* __restrict__ W, __bf16* __restrict__ Wt)
{
    const int idx = blockIdx.x * 1024 + threadIdx.x;   // over col*FIN+k, coalesced in k
    #pragma unroll
    for (int t = 0; t < 4; ++t) {
        const int e   = idx + t * 256;
        const int col = e >> 9;           // /FIN
        const int k   = e & (FIN - 1);
        Wt[e] = (__bf16)W[(size_t)k * FOUT + col];
    }
}

// ---------------------------------------------------------------------------
// Kernel 1: Wh = h @ W  (bf16 WMMA, f32 accumulate), fused epilogue:
//   - Whbt: bf16 Wh transposed per-batch [b][col][i]  (one b128 store/lane)
//   - Wh1/Wh2 += Wh-tile . a1/a2  (shfl-tree + native global f32 atomics)
// 512 threads = 16 waves; block = 16 rows x 256 cols; wave w -> col tile w.
// ---------------------------------------------------------------------------
__global__ __launch_bounds__(512)
void proj_gemm(const float* __restrict__ h, const __bf16* __restrict__ Wt,
               const float* __restrict__ avec,
               __bf16* __restrict__ Whbt,
               float* __restrict__ Wh1, float* __restrict__ Wh2)
{
    const int wave = threadIdx.x >> 5;
    const int lane = threadIdx.x & 31;
    const int half = lane >> 4;          // 0: lanes 0-15, 1: lanes 16-31
    const int l16  = lane & 15;
    const int row0 = blockIdx.x * 16;    // global row (b*2048 + i)
    const int col0 = wave * 16;

    v8f acc = {};
    for (int k0 = 0; k0 < FIN; k0 += 32) {
        // A fragment 16x32 bf16: lane row=l16; K = half*16+{0..15}
        v16bf a;
        const float* ap = h + (size_t)(row0 + l16) * FIN + k0 + half * 16;
        #pragma unroll
        for (int kk = 0; kk < 8; ++kk) {
            a[kk]     = (__bf16)ap[kk];
            a[8 + kk] = (__bf16)ap[8 + kk];
        }
        // B fragment 32x16 bf16: lane col=l16; K = half*16 + v, contiguous in Wt
        const v16bf b = *(const v16bf*)(Wt + (size_t)(col0 + l16) * FIN + k0 + half * 16);
        acc = __builtin_amdgcn_wmma_f32_16x16x32_bf16(false, a, false, b,
                                                      (short)0, acc, false, false);
    }

    // --- packed bf16 store: acc[i] -> row half*8+i, fixed col => contiguous
    const int c   = col0 + l16;
    const int b   = row0 >> 11;                  // batch
    const int ri0 = (row0 & (NN - 1)) + half * 8;
    v8bf pk;
    #pragma unroll
    for (int i = 0; i < 8; ++i) pk[i] = (__bf16)acc[i];
    *(v8bf*)(Whbt + ((size_t)b * FOUT + c) * NN + ri0) = pk;

    // --- fused attention-vector projections: Wh1/Wh2 partial dots
    const float a1 = avec[c];
    const float a2 = avec[FOUT + c];
    float r1[8], r2[8];
    #pragma unroll
    for (int i = 0; i < 8; ++i) { r1[i] = acc[i] * a1; r2[i] = acc[i] * a2; }
    #pragma unroll
    for (int off = 1; off < 16; off <<= 1) {
        #pragma unroll
        for (int i = 0; i < 8; ++i) {
            r1[i] += __shfl_xor(r1[i], off, 32);
            r2[i] += __shfl_xor(r2[i], off, 32);
        }
    }
    if (l16 == 0) {
        #pragma unroll
        for (int i = 0; i < 8; ++i) {
            unsafeAtomicAdd(&Wh1[row0 + half * 8 + i], r1[i]);
            unsafeAtomicAdd(&Wh2[row0 + half * 8 + i], r2[i]);
        }
    }
}

// ---------------------------------------------------------------------------
// Kernel 2: masked online-softmax row stats (max m, sum l). One wave per row.
// ---------------------------------------------------------------------------
__global__ __launch_bounds__(256)
void row_stats(const int* __restrict__ adj, const float* __restrict__ Wh1,
               const float* __restrict__ Wh2,
               float* __restrict__ rowM, float* __restrict__ rowL)
{
    const int r    = blockIdx.x * 8 + (threadIdx.x >> 5);
    const int lane = threadIdx.x & 31;
    const int b    = r >> 11;
    const float e1 = Wh1[r];
    const int*   arow = adj + (size_t)r * NN;
    const float* w2   = Wh2 + (size_t)b * NN;

    float m = -3.0e38f, l = 0.f;
    for (int j = lane; j < NN; j += 32) {
        __builtin_prefetch(arow + j + 256, 0, 1);
        if (arow[j] > 0) {
            float e = e1 + w2[j];
            e = e > 0.f ? e : ALPHA * e;
            if (e > m) { l = l * __expf(m - e) + 1.f; m = e; }
            else       { l += __expf(e - m); }
        }
    }
    #pragma unroll
    for (int off = 16; off; off >>= 1) {
        const float mo = __shfl_xor(m, off, 32);
        const float lo = __shfl_xor(l, off, 32);
        const float mn = fmaxf(m, mo);
        l = l * __expf(m - mn) + lo * __expf(mo - mn);
        m = mn;
    }
    if (lane == 0) { rowM[r] = m; rowL[r] = l; }
}

// ---------------------------------------------------------------------------
// Kernel 3: O[16,256] = softmax(mask(e)) @ Wh, then ELU. Flash-style.
// 512 threads = 16 waves. cg = wave>>3 selects cols [cg*128, +128),
// wj = wave&7 strides over j-chunks of 32. Probabilities built in-register
// as the bf16 A-fragment; Wh^T bf16 tiles are contiguous B-fragment loads.
// Cross-wave reduce via LDS ds_add_f32, then 1/l scale + ELU.
// ---------------------------------------------------------------------------
__global__ __launch_bounds__(512)
void attn_apply(const int* __restrict__ adj, const __bf16* __restrict__ Whbt,
                const float* __restrict__ Wh1, const float* __restrict__ Wh2,
                const float* __restrict__ rowM, const float* __restrict__ rowL,
                float* __restrict__ out)
{
    __shared__ float lds[16 * FOUT];

    const int tid  = threadIdx.x;
    const int wave = tid >> 5;
    const int lane = tid & 31;
    const int half = lane >> 4;
    const int l16  = lane & 15;
    const int cg   = wave >> 3;          // col group 0/1
    const int wj   = wave & 7;           // j-chunk stride slot
    const int row0 = blockIdx.x * 16;    // global rows
    const int b    = row0 >> 11;
    const int cbase = cg * 128;

    #pragma unroll
    for (int t = tid; t < 16 * FOUT; t += 512) lds[t] = 0.f;
    __syncthreads();

    const int row_g = row0 + l16;        // row this lane's A-fragment covers
    const float e1  = Wh1[row_g];
    const float m   = rowM[row_g];
    const int*   arow = adj + (size_t)row_g * NN;
    const float* w2   = Wh2 + (size_t)b * NN;

    v8f acc[8];
    #pragma unroll
    for (int ct = 0; ct < 8; ++ct) acc[ct] = (v8f){};

    for (int t = 0; t < NN / 32 / 8; ++t) {           // 8 chunks per wave
        const int j0 = (t * 8 + wj) * 32;
        __builtin_prefetch(arow + j0 + 256, 0, 1);
        // A fragment: P[16 rows, 32 j] in bf16. lane row=l16,
        // K(=j-j0) = half*16+{0..15}
        const int jb = j0 + half * 16;
        v16bf pa;
        #pragma unroll
        for (int kk = 0; kk < 8; ++kk) {
            float p0 = 0.f, p1 = 0.f;
            if (arow[jb + kk] > 0) {
                float e = e1 + w2[jb + kk];
                e = e > 0.f ? e : ALPHA * e;
                p0 = __expf(e - m);
            }
            if (arow[jb + 8 + kk] > 0) {
                float e = e1 + w2[jb + 8 + kk];
                e = e > 0.f ? e : ALPHA * e;
                p1 = __expf(e - m);
            }
            pa[kk]     = (__bf16)p0;
            pa[8 + kk] = (__bf16)p1;
        }
        // 8 col tiles: B fragment = Wh^T[col, j0+half*16 .. +16) contiguous
        #pragma unroll
        for (int ct = 0; ct < 8; ++ct) {
            const int col = cbase + ct * 16 + l16;
            const __bf16* bp = Whbt + ((size_t)b * FOUT + col) * NN + j0 + half * 16;
            const v16bf bfrag = *(const v16bf*)bp;
            acc[ct] = __builtin_amdgcn_wmma_f32_16x16x32_bf16(false, pa, false, bfrag,
                                                              (short)0, acc[ct], false, false);
        }
    }

    // cross-wave reduction in LDS (ds_add_f32)
    #pragma unroll
    for (int ct = 0; ct < 8; ++ct) {
        const int col = cbase + ct * 16 + l16;
        #pragma unroll
        for (int i = 0; i < 8; ++i)
            atomicAdd(&lds[(half * 8 + i) * FOUT + col], acc[ct][i]);
    }
    __syncthreads();

    // scale by 1/l, ELU, store
    #pragma unroll
    for (int t = tid; t < 16 * FOUT; t += 512) {
        const int r = t >> 8;            // 0..15
        const int c = t & (FOUT - 1);
        const float v = lds[t] / rowL[row0 + r];
        out[(size_t)(row0 + r) * FOUT + c] = v > 0.f ? v : (__expf(v) - 1.0f);
    }
}

// ---------------------------------------------------------------------------
extern "C" void kernel_launch(void* const* d_in, const int* in_sizes, int n_in,
                              void* d_out, int out_size, void* d_ws, size_t ws_size,
                              hipStream_t stream)
{
    const float* h   = (const float*)d_in[0];   // [8,2048,512]
    const int*   adj = (const int*)d_in[1];     // [8,2048,2048]
    const float* W   = (const float*)d_in[2];   // [512,256]
    const float* a   = (const float*)d_in[3];   // [512]
    float* out = (float*)d_out;                 // [8,2048,256]

    // workspace layout (~8.9 MB)
    __bf16* Whbt = (__bf16*)d_ws;                                 // 16384*256 bf16 (transposed/batch)
    float*  Wh1  = (float*)((char*)Whbt + (size_t)NROWS * FOUT * 2);
    float*  Wh2  = Wh1 + NROWS;
    float*  rowM = Wh2 + NROWS;
    float*  rowL = rowM + NROWS;
    __bf16* Wt   = (__bf16*)(rowL + NROWS);                       // 256*512 bf16 transposed weights

    zero_init <<<(2 * NROWS) / 1024, 256, 0, stream>>>(Wh1, 2 * NROWS); // Wh1+Wh2 contiguous
    prep_wt   <<<(FOUT * FIN) / 1024, 256, 0, stream>>>(W, Wt);
    proj_gemm <<<NROWS / 16, 512, 0, stream>>>(h, Wt, a, Whbt, Wh1, Wh2);
    row_stats <<<NROWS / 8,  256, 0, stream>>>(adj, Wh1, Wh2, rowM, rowL);
    attn_apply<<<NROWS / 16, 512, 0, stream>>>(adj, Whbt, Wh1, Wh2, rowM, rowL, out);
}
